// TinyTribeV3Sparse_14431090115246
// MI455X (gfx1250) — matrix operand order repaired and checked by reading, not compiled, not run on hardware
//
#include <hip/hip_runtime.h>
#include <cstddef>
#include <cstdint>

#define D_MODEL 1024
#define FF_DIM  2048
#define N_EXP   6
#define B_SZ    8
#define T_LEN   2048
#define N_TOK   (B_SZ * T_LEN)     /* 16384 */
#define NFREQ   (T_LEN / 2 + 1)    /* 1025  */
#define M_F     (B_SZ * NFREQ)     /* 8200  */

#define BM 128
#define BN 128
#define BK 32

typedef __attribute__((ext_vector_type(16))) __bf16        bf16x16;
typedef __attribute__((ext_vector_type(8)))  float         f32x8;
typedef __attribute__((ext_vector_type(4)))  unsigned int  u32x4;
typedef __attribute__((ext_vector_type(4)))  unsigned int  v4u;
typedef __attribute__((ext_vector_type(8)))  int           v8i;
typedef __attribute__((ext_vector_type(4)))  int           v4i;

// Tensor Data Mover availability (probe-verified builtin; arity differs by toolchain)
#if defined(__has_builtin)
#if __has_builtin(__builtin_amdgcn_tensor_load_to_lds) && __has_builtin(__builtin_amdgcn_s_wait_tensorcnt)
#define USE_TDM 1
#endif
#endif
#ifndef USE_TDM
#define USE_TDM 0
#endif

union FragAB { bf16x16 v; u32x4 q[2]; };
union Pack16 { u32x4 q[2]; unsigned short h[16]; };

struct Cplx { float x, y; };

__device__ __forceinline__ unsigned short f2bf(float f) {
  unsigned int u = __float_as_uint(f);
  u += 0x7FFFu + ((u >> 16) & 1u);          // round-to-nearest-even
  return (unsigned short)(u >> 16);
}

__device__ __forceinline__ float gelu_exact(float v) {
  return 0.5f * v * (1.0f + erff(v * 0.70710678118654752f));
}

// ---------------------------------------------------------------- casts
__global__ void cast_bf16_kernel(const float* __restrict__ src,
                                 unsigned short* __restrict__ dst, long long n) {
  long long i = (long long)blockIdx.x * blockDim.x + threadIdx.x;
  const long long stride = (long long)gridDim.x * blockDim.x;
  for (; i < n; i += stride) dst[i] = f2bf(src[i]);
}

// ---------------------------------------------------------------- router
__global__ __launch_bounds__(256) void router_kernel(
    const float* __restrict__ x, const float* __restrict__ rw, const float* __restrict__ rb,
    const float* __restrict__ gw, const float* __restrict__ gb,
    float* __restrict__ gates, float* __restrict__ probsbuf, float* __restrict__ sig)
{
  const int token = blockIdx.x * 8 + (threadIdx.x >> 5);
  const int lane  = threadIdx.x & 31;
  const float* xr = x + (size_t)token * D_MODEL;
  float acc[N_EXP] = {0.f, 0.f, 0.f, 0.f, 0.f, 0.f};
  float ag = 0.f;
  for (int k = lane; k < D_MODEL; k += 32) {
    const float xv = xr[k];
#pragma unroll
    for (int e = 0; e < N_EXP; ++e) acc[e] += xv * rw[k * N_EXP + e];
    ag += xv * gw[k];
  }
#pragma unroll
  for (int off = 16; off >= 1; off >>= 1) {
#pragma unroll
    for (int e = 0; e < N_EXP; ++e) acc[e] += __shfl_down(acc[e], off, 32);
    ag += __shfl_down(ag, off, 32);
  }
  if (lane == 0) {
    float lg[N_EXP], mx = -1e30f;
#pragma unroll
    for (int e = 0; e < N_EXP; ++e) { lg[e] = acc[e] + rb[e]; mx = fmaxf(mx, lg[e]); }
    float s = 0.f;
#pragma unroll
    for (int e = 0; e < N_EXP; ++e) { lg[e] = expf(lg[e] - mx); s += lg[e]; }
    const float inv = 1.f / s;
    float p[N_EXP];
#pragma unroll
    for (int e = 0; e < N_EXP; ++e) p[e] = lg[e] * inv;
    int i1 = 0;
#pragma unroll
    for (int e = 1; e < N_EXP; ++e) if (p[e] > p[i1]) i1 = e;
    int i2 = (i1 == 0) ? 1 : 0;
#pragma unroll
    for (int e = 0; e < N_EXP; ++e) if (e != i1 && p[e] > p[i2]) i2 = e;
#pragma unroll
    for (int e = 0; e < N_EXP; ++e) {
      probsbuf[(size_t)token * N_EXP + e] = p[e];
      gates[(size_t)token * N_EXP + e] =
          (e == i1) ? p[i1] : ((e == i2) ? p[i2] : 0.f);
    }
    sig[token] = 1.f / (1.f + expf(-(ag + gb[0])));
  }
}

// ---------------------------------------------------------------- aux loss (deterministic tree reduce)
__global__ __launch_bounds__(256) void aux_kernel(const float* __restrict__ probsbuf,
                                                  const float* __restrict__ gates,
                                                  float* __restrict__ aux_out)
{
  __shared__ float sImp[N_EXP][256];
  __shared__ float sLoad[N_EXP][256];
  const int tid = threadIdx.x;
  float impL[N_EXP] = {0.f, 0.f, 0.f, 0.f, 0.f, 0.f};
  float loadL[N_EXP] = {0.f, 0.f, 0.f, 0.f, 0.f, 0.f};
  for (int t = tid; t < N_TOK; t += 256) {
#pragma unroll
    for (int e = 0; e < N_EXP; ++e) {
      impL[e]  += probsbuf[(size_t)t * N_EXP + e];
      loadL[e] += (gates[(size_t)t * N_EXP + e] > 0.f) ? 1.f : 0.f;
    }
  }
#pragma unroll
  for (int e = 0; e < N_EXP; ++e) { sImp[e][tid] = impL[e]; sLoad[e][tid] = loadL[e]; }
  __syncthreads();
  for (int off = 128; off >= 1; off >>= 1) {
    if (tid < off) {
#pragma unroll
      for (int e = 0; e < N_EXP; ++e) {
        sImp[e][tid]  += sImp[e][tid + off];
        sLoad[e][tid] += sLoad[e][tid + off];
      }
    }
    __syncthreads();
  }
  if (tid == 0) {
    float s = 0.f;
#pragma unroll
    for (int e = 0; e < N_EXP; ++e)
      s += (sLoad[e][0] / ((float)N_TOK * 2.f)) * (sImp[e][0] / (float)N_TOK);
    aux_out[0] = (float)N_EXP * s;
  }
}

// ---------------------------------------------------------------- conv prep: xc = bf16(x + depthwise_conv(x))
__global__ void conv_prep_kernel(const float* __restrict__ x,
                                 const float* __restrict__ k3,
                                 unsigned short* __restrict__ xc)
{
  const long long i = (long long)blockIdx.x * blockDim.x + threadIdx.x;
  if (i >= (long long)N_TOK * D_MODEL) return;
  const int d = (int)(i % D_MODEL);
  const long long bt = i / D_MODEL;
  const int t = (int)(bt % T_LEN);
  const long long bBase = (bt - t) * D_MODEL;   // b*T*D
  float c = 0.f;
#pragma unroll
  for (int j = 0; j < 3; ++j) {
    const int tt = t - 1 + j;
    if (tt >= 0 && tt < T_LEN)
      c += x[bBase + (long long)tt * D_MODEL + d] * k3[d * 3 + j];
  }
  xc[i] = f2bf(x[i] + c);
}

// ---------------------------------------------------------------- WMMA GEMM (bf16 A/W, f32 acc)
// mode 0: Hout[M,N] = bf16(gelu(A@W + bias))
// mode 1: Fout[M,N] (+)= gate[row]*(A@W + bias)   (gate==nullptr -> 1.0)
__global__ __launch_bounds__(256) void gemm_wmma_kernel(
    const unsigned short* __restrict__ A,   // [M,K] bf16
    const unsigned short* __restrict__ W,   // [K,N] bf16
    const float* __restrict__ bias,         // [N]
    unsigned short* __restrict__ Hout,
    float* __restrict__ Fout,
    const float* __restrict__ gate, int gstride, int goff,
    int mode, int accflag, int M, int N, int K)
{
  __shared__ unsigned short sA[BM][BK + 8];   // 16B-aligned row pitch (80B = 20 DW)
  __shared__ unsigned short sB[BN][BK + 8];   // B stored transposed: [n][k]
  const int tid  = threadIdx.x;
  const int wid  = tid >> 5;
  const int lane = tid & 31;
  const int lr   = lane & 15;
  const int lh   = lane >> 4;
  const int m0 = blockIdx.y * BM;
  const int n0 = blockIdx.x * BN;
  const int wm = (wid >> 2) * 64;   // 2 waves along M
  const int wn = (wid & 3) * 32;    // 4 waves along N

  f32x8 acc[4][2];
  const f32x8 zero = {0.f, 0.f, 0.f, 0.f, 0.f, 0.f, 0.f, 0.f};
#pragma unroll
  for (int mi = 0; mi < 4; ++mi)
#pragma unroll
    for (int ni = 0; ni < 2; ++ni) acc[mi][ni] = zero;

  const int arow = tid >> 1;
  const int acol = (tid & 1) * 16;
  const int bk   = tid >> 3;
  const int bn   = (tid & 7) * 16;

  for (int k0 = 0; k0 < K; k0 += BK) {
#if USE_TDM
    // ---- Tensor Data Mover: DMA the 128x32 bf16 A tile straight into LDS.
    // D# group0: count=1 | lds_addr | global_addr | type=2
    // D# group1: data_size=2B, pad_enable (16 DW data + 4 DW pad == 80B LDS row pitch),
    //            tensor = [dim0=K, dim1=rows-left] stride K, tile = 32 x 128.
    // OOB rows (tail M-block) are zero-filled by the TDM per spec.
    if (wid == 0) {
      const unsigned long long ga =
          (unsigned long long)(uintptr_t)(A + (size_t)m0 * K + k0);
      const unsigned lds_a = (unsigned)(uintptr_t)&sA[0][0];
      const unsigned rows = (unsigned)((M - m0) < BM ? (M - m0) : BM);
      v4u g0;
      g0[0] = 1u;                                                   // count=1
      g0[1] = lds_a;                                                // lds_addr
      g0[2] = (unsigned)ga;                                         // global_addr[31:0]
      g0[3] = ((unsigned)(ga >> 32) & 0x01FFFFFFu) | (2u << 30);    // addr[56:32] | type=2
      v8i g1;
      g1[0] = (int)((1u << 16) | (1u << 20) | (3u << 22) | (3u << 25)); // 2B, pad 16DW->+4DW
      g1[1] = (int)(((unsigned)K & 0xFFFFu) << 16);                 // tensor_dim0 lo16
      g1[2] = (int)((((unsigned)K >> 16) & 0xFFFFu) | ((rows & 0xFFFFu) << 16));
      g1[3] = (int)(((rows >> 16) & 0xFFFFu) | ((unsigned)BK << 16)); // dim1 hi | tile_dim0
      g1[4] = (int)((unsigned)BM);                                  // tile_dim1=128, tile_dim2=0
      g1[5] = (int)K;                                               // tensor_dim0_stride lo32
      g1[6] = 0;                                                    // stride hi | dim1_stride lo
      g1[7] = 0;
      const v4i gz = {0, 0, 0, 0};
#if __clang_major__ >= 23
      const v8i gz8 = {0, 0, 0, 0, 0, 0, 0, 0};
      __builtin_amdgcn_tensor_load_to_lds(g0, g1, gz, gz, gz8, 0);
#else
      __builtin_amdgcn_tensor_load_to_lds(g0, g1, gz, gz, 0);
#endif
      __builtin_amdgcn_s_wait_tensorcnt(0);
    }
#else
    { // manual stage A block 128x32
      u32x4 v0 = {0, 0, 0, 0}, v1 = {0, 0, 0, 0};
      if (m0 + arow < M) {
        const unsigned short* gp = A + (size_t)(m0 + arow) * K + k0 + acol;
        v0 = *(const u32x4*)gp;
        v1 = *(const u32x4*)(gp + 8);
      }
      *(u32x4*)&sA[arow][acol]     = v0;
      *(u32x4*)&sA[arow][acol + 8] = v1;
    }
#endif
    { // stage B block 32x128, transposed into LDS (TDM has no transpose path)
      const unsigned short* gp = W + (size_t)(k0 + bk) * N + n0 + bn;
      Pack16 u;
      u.q[0] = *(const u32x4*)gp;
      u.q[1] = *(const u32x4*)(gp + 8);
#pragma unroll
      for (int i = 0; i < 16; ++i) sB[bn + i][bk] = u.h[i];
    }
    __syncthreads();

    if (k0 + BK < K) {  // global_prefetch_b8 of next K tile of W
      __builtin_prefetch(W + (size_t)(k0 + BK + bk) * N + n0 + bn, 0, 1);
#if !USE_TDM
      __builtin_prefetch(A + (size_t)(m0 + arow) * K + (k0 + BK) + acol, 0, 1);
#endif
    }

    // A fragments: lanes 0-15 hold K0..7 / K16..23, lanes 16-31 hold K8..15 / K24..31
    FragAB afr[4];
#pragma unroll
    for (int mi = 0; mi < 4; ++mi) {
      const int mrow = wm + mi * 16 + lr;
      afr[mi].q[0] = *(const u32x4*)&sA[mrow][lh * 8];
      afr[mi].q[1] = *(const u32x4*)&sA[mrow][16 + lh * 8];
    }
#pragma unroll
    for (int ni = 0; ni < 2; ++ni) {
      // B fragment: lane = column, lanes 0-15 K0..7 / K16..23, lanes 16-31 K8..15 / K24..31
      FragAB bfr;
      const int nrow = wn + ni * 16 + lr;
      bfr.q[0] = *(const u32x4*)&sB[nrow][lh * 8];
      bfr.q[1] = *(const u32x4*)&sB[nrow][16 + lh * 8];
#pragma unroll
      for (int mi = 0; mi < 4; ++mi) {
        acc[mi][ni] = __builtin_amdgcn_wmma_f32_16x16x32_bf16(
            false, afr[mi].v, false, bfr.v, (short)0, acc[mi][ni], false, false);
      }
    }
    __syncthreads();
  }

  // epilogue: C layout — VGPR r holds M = r + 8*(lane>>4), N = lane&15
#pragma unroll
  for (int mi = 0; mi < 4; ++mi) {
#pragma unroll
    for (int ni = 0; ni < 2; ++ni) {
#pragma unroll
      for (int r = 0; r < 8; ++r) {
        const int row = m0 + wm + mi * 16 + lh * 8 + r;
        const int col = n0 + wn + ni * 16 + lr;
        if (row < M) {
          float v = acc[mi][ni][r] + bias[col];
          const size_t o = (size_t)row * N + col;
          if (mode == 0) {
            Hout[o] = f2bf(gelu_exact(v));
          } else {
            const float s = gate ? gate[(size_t)row * gstride + goff] : 1.0f;
            if (accflag) Fout[o] += s * v; else Fout[o] = s * v;
          }
        }
      }
    }
  }
}

// ---------------------------------------------------------------- 2048-pt Stockham FFT in LDS
__device__ Cplx* fft2048_lds(Cplx* s, Cplx* d, float sign) {
  int l = T_LEN >> 1, ms = 0;
  while (l >= 1) {
    const int m = 1 << ms;
    for (int i = threadIdx.x; i < (T_LEN >> 1); i += 256) {
      const int j = i >> ms, k = i & (m - 1);
      const Cplx c0 = s[k + j * m];
      const Cplx c1 = s[k + j * m + l * m];
      const float ang = sign * 3.14159265358979323846f * (float)j / (float)l;
      float sn, cs;
      __sincosf(ang, &sn, &cs);
      Cplx e0; e0.x = c0.x + c1.x; e0.y = c0.y + c1.y;
      d[k + 2 * j * m] = e0;
      const float dr = c0.x - c1.x, di = c0.y - c1.y;
      Cplx e1; e1.x = dr * cs - di * sn; e1.y = dr * sn + di * cs;
      d[k + 2 * j * m + m] = e1;
    }
    __syncthreads();
    Cplx* t = s; s = d; d = t;
    l >>= 1; ++ms;
  }
  return s;
}

__global__ __launch_bounds__(256) void fft_fwd_kernel(const float* __restrict__ x,
                                                      unsigned short* __restrict__ Af)
{
  __shared__ Cplx bufA[T_LEN];
  __shared__ Cplx bufB[T_LEN];
  const int col = blockIdx.x;
  const int b = col / D_MODEL, d = col % D_MODEL;
  const float* xp = x + (size_t)b * T_LEN * D_MODEL + d;
  for (int t = threadIdx.x; t < T_LEN; t += 256) {
    bufA[t].x = xp[(size_t)t * D_MODEL];
    bufA[t].y = 0.f;
  }
  __syncthreads();
  Cplx* res = fft2048_lds(bufA, bufB, -1.0f);
  for (int t = threadIdx.x; t < NFREQ; t += 256) {
    const size_t r = (size_t)(b * NFREQ + t) * (2 * D_MODEL);
    Af[r + d]           = f2bf(res[t].x);
    Af[r + D_MODEL + d] = f2bf(res[t].y);
  }
}

__global__ __launch_bounds__(256) void ifft_acc_kernel(const float* __restrict__ FO,
                                                       const float* __restrict__ gates, int eid,
                                                       float* __restrict__ out)
{
  __shared__ Cplx bufA[T_LEN];
  __shared__ Cplx bufB[T_LEN];
  const int col = blockIdx.x;
  const int b = col / D_MODEL, d = col % D_MODEL;
  for (int t = threadIdx.x; t < T_LEN; t += 256) {  // Hermitian extension
    const int ts = (t <= T_LEN / 2) ? t : (T_LEN - t);
    const size_t r = (size_t)(b * NFREQ + ts) * (2 * D_MODEL);
    float re = FO[r + d];
    float im = FO[r + D_MODEL + d];
    if (t > T_LEN / 2) im = -im;
    bufA[t].x = re; bufA[t].y = im;
  }
  __syncthreads();
  Cplx* res = fft2048_lds(bufA, bufB, +1.0f);
  const float invT = 1.0f / (float)T_LEN;
  for (int t = threadIdx.x; t < T_LEN; t += 256) {
    const size_t token = (size_t)b * T_LEN + t;
    out[token * D_MODEL + d] += gates[token * N_EXP + eid] * res[t].x * invT;
  }
}

// ---------------------------------------------------------------- host orchestration
extern "C" void kernel_launch(void* const* d_in, const int* in_sizes, int n_in,
                              void* d_out, int out_size, void* d_ws, size_t ws_size,
                              hipStream_t stream)
{
  (void)in_sizes; (void)n_in; (void)out_size; (void)ws_size;
  const float* x   = (const float*)d_in[0];
  const float* sw1 = (const float*)d_in[1];
  const float* sb1 = (const float*)d_in[2];
  const float* sw2 = (const float*)d_in[3];
  const float* sb2 = (const float*)d_in[4];
  const float* gw  = (const float*)d_in[5];
  const float* gb  = (const float*)d_in[6];
  const float* rw  = (const float*)d_in[7];
  const float* rb  = (const float*)d_in[8];
  const float* ck  = (const float*)d_in[9];
  const float* cw1 = (const float*)d_in[10];
  const float* cb1 = (const float*)d_in[11];
  const float* cw2 = (const float*)d_in[12];
  const float* cb2 = (const float*)d_in[13];
  const float* fw1 = (const float*)d_in[14];
  const float* fb1 = (const float*)d_in[15];
  const float* fw2 = (const float*)d_in[16];
  const float* fb2 = (const float*)d_in[17];
  const float* mw1 = (const float*)d_in[18];
  const float* mb1 = (const float*)d_in[19];
  const float* mw2 = (const float*)d_in[20];
  const float* mb2 = (const float*)d_in[21];

  float* out = (float*)d_out;
  float* aux = out + (size_t)N_TOK * D_MODEL;

  char* base = (char*)d_ws;
  size_t off = 0;
  auto carve = [&](size_t bytes) -> void* {
    void* p = (void*)(base + off);
    off += (bytes + 255) & ~(size_t)255;
    return p;
  };

  unsigned short* xb     = (unsigned short*)carve((size_t)N_TOK * D_MODEL * 2);
  unsigned short* sw1b   = (unsigned short*)carve((size_t)D_MODEL * FF_DIM * 2);
  unsigned short* sw2b   = (unsigned short*)carve((size_t)FF_DIM * D_MODEL * 2);
  unsigned short* cw1b   = (unsigned short*)carve((size_t)2 * D_MODEL * FF_DIM * 2);
  unsigned short* cw2b   = (unsigned short*)carve((size_t)2 * FF_DIM * D_MODEL * 2);
  unsigned short* fw1b   = (unsigned short*)carve((size_t)2 * 2 * D_MODEL * FF_DIM * 2);
  unsigned short* fw2b   = (unsigned short*)carve((size_t)2 * FF_DIM * 2 * D_MODEL * 2);
  unsigned short* mw1b   = (unsigned short*)carve((size_t)2 * D_MODEL * FF_DIM * 2);
  unsigned short* mw2b   = (unsigned short*)carve((size_t)2 * FF_DIM * D_MODEL * 2);
  unsigned short* hidden = (unsigned short*)carve((size_t)N_TOK * FF_DIM * 2);
  unsigned short* Af     = (unsigned short*)carve((size_t)M_F * 2 * D_MODEL * 2);
  unsigned short* xc     = (unsigned short*)carve((size_t)N_TOK * D_MODEL * 2);
  float* freq  = (float*)carve((size_t)M_F * 2 * D_MODEL * 4);
  float* gates = (float*)carve((size_t)N_TOK * N_EXP * 4);
  float* probs = (float*)carve((size_t)N_TOK * N_EXP * 4);
  float* sig   = (float*)carve((size_t)N_TOK * 4);

  // --- bf16 casts (weights become L2-resident working set; A matrices halve HBM traffic)
  cast_bf16_kernel<<<2048, 256, 0, stream>>>(x,   xb,   (long long)N_TOK * D_MODEL);
  cast_bf16_kernel<<<2048, 256, 0, stream>>>(sw1, sw1b, (long long)D_MODEL * FF_DIM);
  cast_bf16_kernel<<<2048, 256, 0, stream>>>(sw2, sw2b, (long long)FF_DIM * D_MODEL);
  cast_bf16_kernel<<<2048, 256, 0, stream>>>(cw1, cw1b, (long long)2 * D_MODEL * FF_DIM);
  cast_bf16_kernel<<<2048, 256, 0, stream>>>(cw2, cw2b, (long long)2 * FF_DIM * D_MODEL);
  cast_bf16_kernel<<<2048, 256, 0, stream>>>(fw1, fw1b, (long long)2 * 2 * D_MODEL * FF_DIM);
  cast_bf16_kernel<<<2048, 256, 0, stream>>>(fw2, fw2b, (long long)2 * FF_DIM * 2 * D_MODEL);
  cast_bf16_kernel<<<2048, 256, 0, stream>>>(mw1, mw1b, (long long)2 * D_MODEL * FF_DIM);
  cast_bf16_kernel<<<2048, 256, 0, stream>>>(mw2, mw2b, (long long)2 * FF_DIM * D_MODEL);

  // --- router, gate, aux loss
  router_kernel<<<N_TOK / 8, 256, 0, stream>>>(x, rw, rb, gw, gb, gates, probs, sig);
  aux_kernel<<<1, 256, 0, stream>>>(probs, gates, aux);

  // --- shared expert (writes/initializes `out`, sigmoid-gated)
  {
    dim3 g1(FF_DIM / BN, N_TOK / BM);
    gemm_wmma_kernel<<<g1, 256, 0, stream>>>(xb, sw1b, sb1, hidden, nullptr,
                                             nullptr, 0, 0, 0, 0, N_TOK, FF_DIM, D_MODEL);
    dim3 g2(D_MODEL / BN, N_TOK / BM);
    gemm_wmma_kernel<<<g2, 256, 0, stream>>>(hidden, sw2b, sb2, nullptr, out,
                                             sig, 1, 0, 1, 0, N_TOK, D_MODEL, FF_DIM);
  }

  // --- rfft(x) once (shared by both fourier experts)
  fft_fwd_kernel<<<B_SZ * D_MODEL, 256, 0, stream>>>(x, Af);

  const int MFB = (M_F + BM - 1) / BM;   // 65 row-blocks, tail zero-filled by TDM/guard
  for (int j = 0; j < 2; ++j) {
    // conv expert, eid = 3j
    conv_prep_kernel<<<(N_TOK * D_MODEL) / 256, 256, 0, stream>>>(
        x, ck + (size_t)j * D_MODEL * 3, xc);
    {
      dim3 g1(FF_DIM / BN, N_TOK / BM);
      gemm_wmma_kernel<<<g1, 256, 0, stream>>>(xc, cw1b + (size_t)j * D_MODEL * FF_DIM,
                                               cb1 + (size_t)j * FF_DIM, hidden, nullptr,
                                               nullptr, 0, 0, 0, 0, N_TOK, FF_DIM, D_MODEL);
      dim3 g2(D_MODEL / BN, N_TOK / BM);
      gemm_wmma_kernel<<<g2, 256, 0, stream>>>(hidden, cw2b + (size_t)j * FF_DIM * D_MODEL,
                                               cb2 + (size_t)j * D_MODEL, nullptr, out,
                                               gates, N_EXP, 3 * j, 1, 1, N_TOK, D_MODEL, FF_DIM);
    }
    // fourier expert, eid = 3j+1
    {
      dim3 g1(FF_DIM / BN, MFB);
      gemm_wmma_kernel<<<g1, 256, 0, stream>>>(Af, fw1b + (size_t)j * 2 * D_MODEL * FF_DIM,
                                               fb1 + (size_t)j * FF_DIM, hidden, nullptr,
                                               nullptr, 0, 0, 0, 0, M_F, FF_DIM, 2 * D_MODEL);
      dim3 g2((2 * D_MODEL) / BN, MFB);
      gemm_wmma_kernel<<<g2, 256, 0, stream>>>(hidden, fw2b + (size_t)j * FF_DIM * 2 * D_MODEL,
                                               fb2 + (size_t)j * 2 * D_MODEL, nullptr, freq,
                                               nullptr, 0, 0, 1, 0, M_F, 2 * D_MODEL, FF_DIM);
      ifft_acc_kernel<<<B_SZ * D_MODEL, 256, 0, stream>>>(freq, gates, 3 * j + 1, out);
    }
    // mlp expert, eid = 3j+2
    {
      dim3 g1(FF_DIM / BN, N_TOK / BM);
      gemm_wmma_kernel<<<g1, 256, 0, stream>>>(xb, mw1b + (size_t)j * D_MODEL * FF_DIM,
                                               mb1 + (size_t)j * FF_DIM, hidden, nullptr,
                                               nullptr, 0, 0, 0, 0, N_TOK, FF_DIM, D_MODEL);
      dim3 g2(D_MODEL / BN, N_TOK / BM);
      gemm_wmma_kernel<<<g2, 256, 0, stream>>>(hidden, mw2b + (size_t)j * FF_DIM * D_MODEL,
                                               mb2 + (size_t)j * D_MODEL, nullptr, out,
                                               gates, N_EXP, 3 * j + 2, 1, 1, N_TOK, D_MODEL, FF_DIM);
    }
  }
}
